// GATv2modif_88381837017296
// MI455X (gfx1250) — matrix operand order repaired and checked by reading, not compiled, not run on hardware
//
#include <hip/hip_runtime.h>
#include <hip/hip_bf16.h>
#include <math.h>

#define N_NODES 50000
#define N_EDGES 800000
#define F_IN    128
#define HCDIM   128   // H * HID
#define HIDP    64    // projection hidden
#define NEG_SLOPE 0.2f

typedef __attribute__((ext_vector_type(16))) _Float16 v16h;
typedef __attribute__((ext_vector_type(8)))  _Float16 v8h;
typedef __attribute__((ext_vector_type(8)))  float    v8f;
typedef __attribute__((ext_vector_type(4)))  float    v4f;

// ---------------------------------------------------------------- utilities
__global__ void fill_f32(float* __restrict__ p, float v, int n) {
    int t = blockIdx.x * blockDim.x + threadIdx.x;
    if (t < n) p[t] = v;
}

__global__ void cvt_f16(const float* __restrict__ src, _Float16* __restrict__ dst, int n) {
    int t = blockIdx.x * blockDim.x + threadIdx.x;
    if (t < n) dst[t] = (_Float16)src[t];
}

__global__ void relu_cvt_f16(const float* __restrict__ src, _Float16* __restrict__ dst, int n) {
    int t = blockIdx.x * blockDim.x + threadIdx.x;
    if (t < n) {
        float v = src[t];
        dst[t] = (_Float16)(v > 0.f ? v : 0.f);
    }
}

// ------------------------------------------------- WMMA GEMM: out = A @ W^T + bias
// A: [M, K] f16 row-major, W: [Nout, K] f16 row-major, out: [M, Nout] f32.
// One wave per 16x16 output tile, K stepped by 32 (v_wmma_f32_16x16x32_f16).
__global__ __launch_bounds__(128)
void gemm_wmma_f16(const _Float16* __restrict__ A, const _Float16* __restrict__ W,
                   const float* __restrict__ bias, float* __restrict__ out,
                   int M, int K, int Nout) {
    const int ntn    = Nout >> 4;
    const int ntiles = (M >> 4) * ntn;
    const int tile   = blockIdx.x * (blockDim.x >> 5) + (threadIdx.x >> 5);
    if (tile >= ntiles) return;            // wave-uniform: EXEC stays all-ones for WMMA
    const int tm   = tile / ntn;
    const int tn   = tile % ntn;
    const int lane = threadIdx.x & 31;
    const int half = lane >> 4;            // K-group select per ISA A/B layouts
    const int l16  = lane & 15;

    const _Float16* arow = A + (size_t)(tm * 16 + l16) * K;   // row M = l16
    const _Float16* brow = W + (size_t)(tn * 16 + l16) * K;   // col N = l16 (B = W^T)

    v8f acc = {};
#pragma unroll
    for (int kb = 0; kb < 128; kb += 32) {
        // A lane data: K = kb + half*8 + [0..7]  and  K = kb + 16 + half*8 + [0..7]
        v8h a0 = *(const v8h*)(arow + kb + half * 8);
        v8h a1 = *(const v8h*)(arow + kb + 16 + half * 8);
        // B lane data: K = kb + half*8 + [0..7]  and  K = kb + 16 + half*8 + [0..7] (same as A)
        v8h b0 = *(const v8h*)(brow + kb + half * 8);
        v8h b1 = *(const v8h*)(brow + kb + 16 + half * 8);
        v16h a, b;
#pragma unroll
        for (int i = 0; i < 8; ++i) {
            a[i] = a0[i]; a[i + 8] = a1[i];
            b[i] = b0[i]; b[i + 8] = b1[i];
        }
        acc = __builtin_amdgcn_wmma_f32_16x16x32_f16(
                  false, a, false, b, (short)0, acc, false, false);
    }

    const int   col = tn * 16 + l16;
    const float bc  = bias[col];
#pragma unroll
    for (int r = 0; r < 8; ++r) {
        const int row = tm * 16 + r + half * 8;   // D layout: M = r (+8 for lanes 16..31)
        out[(size_t)row * Nout + col] = acc[r] + bc;
    }
}

// ------------------------------------------------- edge phase: raw attention logits
// One wave per edge; lane handles 4 channels (lanes 0-15 = head 0, 16-31 = head 1).
__global__ __launch_bounds__(256)
void edge_alpha(const float* __restrict__ xw, const float* __restrict__ att,
                const int* __restrict__ ei, float* __restrict__ alpha, int E) {
    const int wave = (blockIdx.x * blockDim.x + threadIdx.x) >> 5;
    const int lane = threadIdx.x & 31;
    if (wave >= E) return;
    const int src = ei[wave];
    const int dst = ei[E + wave];
    const int c0  = lane * 4;
    v4f xs = *(const v4f*)(xw + (size_t)src * HCDIM + c0);
    v4f xd = *(const v4f*)(xw + (size_t)dst * HCDIM + c0);
    v4f av = *(const v4f*)(att + c0);
    float p = 0.f;
#pragma unroll
    for (int i = 0; i < 4; ++i) {
        float v = xs[i] + xd[i];
        v = v > 0.f ? v : NEG_SLOPE * v;   // leaky_relu
        p += v * av[i];
    }
    // reduce over the 16 lanes of this head (xor masks 1,2,4,8 stay in the half-wave)
#pragma unroll
    for (int m = 1; m < 16; m <<= 1) p += __shfl_xor(p, m, 32);
    if ((lane & 15) == 0) alpha[(size_t)wave * 2 + (lane >> 4)] = p;
}

// segment max (float atomic-max via signed/unsigned int trick; segm init = -inf)
__global__ void seg_max_kernel(const float* __restrict__ alpha, const int* __restrict__ ei,
                               float* __restrict__ segm, int E) {
    int t = blockIdx.x * blockDim.x + threadIdx.x;
    if (t >= E * 2) return;
    int e = t >> 1, h = t & 1;
    int dst = ei[E + e];
    float v = alpha[t];
    int* ai = (int*)(segm + (size_t)dst * 2 + h);
    if (v >= 0.f) atomicMax(ai, __float_as_int(v));
    else          atomicMin((unsigned int*)ai, __float_as_uint(v));
}

// ex = exp(alpha - m) (empty-segment guard), accumulate segment sums
__global__ void edge_expsum(float* __restrict__ alpha, const int* __restrict__ ei,
                            const float* __restrict__ segm, float* __restrict__ segs, int E) {
    int t = blockIdx.x * blockDim.x + threadIdx.x;
    if (t >= E * 2) return;
    int e = t >> 1, h = t & 1;
    int dst = ei[E + e];
    float m = segm[(size_t)dst * 2 + h];
    if (!(m >= -3.0e38f)) m = 0.f;        // -inf (empty segment) / NaN guard
    float ex = __expf(alpha[t] - m);
    alpha[t] = ex;
    atomicAdd(segs + (size_t)dst * 2 + h, ex);
}

// scatter normalized messages: agg[dst] += xw[src] * softmax(alpha)
__global__ __launch_bounds__(256)
void scatter_msg(const float* __restrict__ xw, const float* __restrict__ alpha,
                 const float* __restrict__ segs, const int* __restrict__ ei,
                 float* __restrict__ agg, int E) {
    const int wave = (blockIdx.x * blockDim.x + threadIdx.x) >> 5;
    const int lane = threadIdx.x & 31;
    if (wave >= E) return;
    const int src = ei[wave];
    const int dst = ei[E + wave];
    const int h   = lane >> 4;
    const float a = alpha[(size_t)wave * 2 + h] /
                    (segs[(size_t)dst * 2 + h] + 1e-16f);
    const int c0 = lane * 4;
    v4f xs = *(const v4f*)(xw + (size_t)src * HCDIM + c0);
    float* ap = agg + (size_t)dst * HCDIM + c0;
    atomicAdd(ap + 0, xs[0] * a);
    atomicAdd(ap + 1, xs[1] * a);
    atomicAdd(ap + 2, xs[2] * a);
    atomicAdd(ap + 3, xs[3] * a);
}

// final: out = sigmoid(h64 @ Wp2^T + bp2), Wp2 is [1,64]
__global__ void final_proj(const float* __restrict__ h64, const float* __restrict__ Wp2,
                           const float* __restrict__ bp2, float* __restrict__ out, int n) {
    int t = blockIdx.x * blockDim.x + threadIdx.x;
    if (t >= n) return;
    const float* hp = h64 + (size_t)t * HIDP;
    float s = bp2[0];
#pragma unroll 8
    for (int j = 0; j < HIDP; ++j) s += hp[j] * Wp2[j];
    out[t] = 1.f / (1.f + __expf(-s));
}

// ---------------------------------------------------------------- driver
extern "C" void kernel_launch(void* const* d_in, const int* in_sizes, int n_in,
                              void* d_out, int out_size, void* d_ws, size_t ws_size,
                              hipStream_t stream) {
    const float* x    = (const float*)d_in[0];
    const int*   ei   = (const int*)d_in[1];
    const float* W1   = (const float*)d_in[2];
    const float* b1   = (const float*)d_in[3];
    const float* att1 = (const float*)d_in[4];
    const float* W2   = (const float*)d_in[5];
    const float* b2   = (const float*)d_in[6];
    const float* att2 = (const float*)d_in[7];
    const float* Wp1  = (const float*)d_in[8];
    const float* bp1  = (const float*)d_in[9];
    const float* Wp2  = (const float*)d_in[10];
    const float* bp2  = (const float*)d_in[11];
    float* out = (float*)d_out;

    const int N = N_NODES, E = N_EDGES;

    char* ws = (char*)d_ws;
    size_t off = 0;
    auto alloc = [&](size_t bytes) {
        char* p = ws + off;
        off += (bytes + 255) & ~(size_t)255;
        return (void*)p;
    };
    _Float16* x16  = (_Float16*)alloc((size_t)N * HCDIM * sizeof(_Float16));
    _Float16* w16  = (_Float16*)alloc((size_t)HCDIM * F_IN * sizeof(_Float16));
    float*    xw   = (float*)alloc((size_t)N * HCDIM * sizeof(float));
    float*    agg  = (float*)alloc((size_t)N * HCDIM * sizeof(float));
    float*    h64  = (float*)alloc((size_t)N * HIDP * sizeof(float));
    float*    alph = (float*)alloc((size_t)E * 2 * sizeof(float));
    float*    segm = (float*)alloc((size_t)N * 2 * sizeof(float));
    float*    segs = (float*)alloc((size_t)N * 2 * sizeof(float));

    const int TB = 256;
    auto cdiv = [](int a, int b) { return (a + b - 1) / b; };

    const int nfeat    = N * HCDIM;                    // 6.4M
    const int gemmTilesFull = (N / 16) * (HCDIM / 16); // 3125 * 8
    const int gemmTilesProj = (N / 16) * (HIDP / 16);  // 3125 * 4
    const int edgeWaveBlocks = cdiv(E, 8);             // 8 edges (waves) per 256-thr block

    // ---- input conversion
    cvt_f16<<<cdiv(nfeat, TB), TB, 0, stream>>>(x, x16, nfeat);

    // ---- layer 1
    cvt_f16<<<cdiv(HCDIM * F_IN, TB), TB, 0, stream>>>(W1, w16, HCDIM * F_IN);
    gemm_wmma_f16<<<cdiv(gemmTilesFull, 4), 128, 0, stream>>>(x16, w16, b1, xw, N, F_IN, HCDIM);
    edge_alpha<<<edgeWaveBlocks, TB, 0, stream>>>(xw, att1, ei, alph, E);
    fill_f32<<<cdiv(N * 2, TB), TB, 0, stream>>>(segm, -__builtin_huge_valf(), N * 2);
    fill_f32<<<cdiv(N * 2, TB), TB, 0, stream>>>(segs, 0.f, N * 2);
    fill_f32<<<cdiv(nfeat, TB), TB, 0, stream>>>(agg, 0.f, nfeat);
    seg_max_kernel<<<cdiv(E * 2, TB), TB, 0, stream>>>(alph, ei, segm, E);
    edge_expsum<<<cdiv(E * 2, TB), TB, 0, stream>>>(alph, ei, segm, segs, E);
    scatter_msg<<<edgeWaveBlocks, TB, 0, stream>>>(xw, alph, segs, ei, agg, E);
    relu_cvt_f16<<<cdiv(nfeat, TB), TB, 0, stream>>>(agg, x16, nfeat);

    // ---- layer 2
    cvt_f16<<<cdiv(HCDIM * HCDIM, TB), TB, 0, stream>>>(W2, w16, HCDIM * HCDIM);
    gemm_wmma_f16<<<cdiv(gemmTilesFull, 4), 128, 0, stream>>>(x16, w16, b2, xw, N, HCDIM, HCDIM);
    edge_alpha<<<edgeWaveBlocks, TB, 0, stream>>>(xw, att2, ei, alph, E);
    fill_f32<<<cdiv(N * 2, TB), TB, 0, stream>>>(segm, -__builtin_huge_valf(), N * 2);
    fill_f32<<<cdiv(N * 2, TB), TB, 0, stream>>>(segs, 0.f, N * 2);
    fill_f32<<<cdiv(nfeat, TB), TB, 0, stream>>>(agg, 0.f, nfeat);
    seg_max_kernel<<<cdiv(E * 2, TB), TB, 0, stream>>>(alph, ei, segm, E);
    edge_expsum<<<cdiv(E * 2, TB), TB, 0, stream>>>(alph, ei, segm, segs, E);
    scatter_msg<<<edgeWaveBlocks, TB, 0, stream>>>(xw, alph, segs, ei, agg, E);
    relu_cvt_f16<<<cdiv(nfeat, TB), TB, 0, stream>>>(agg, x16, nfeat);

    // ---- projection head: [N,128] @ Wp1^T -> [N,64] (WMMA), then 64 -> 1 + sigmoid
    cvt_f16<<<cdiv(HIDP * HCDIM, TB), TB, 0, stream>>>(Wp1, w16, HIDP * HCDIM);
    gemm_wmma_f16<<<cdiv(gemmTilesProj, 4), 128, 0, stream>>>(x16, w16, bp1, h64, N, HCDIM, HIDP);
    final_proj<<<cdiv(N, TB), TB, 0, stream>>>(h64, Wp2, bp2, out, N);
}
